// NormalizedCausalAttention_62457414419224
// MI455X (gfx1250) — compile-verified, matrix operand order checked by reading, not run on hardware
//
#include <hip/hip_runtime.h>
#include <hip/hip_bf16.h>
#include <stdint.h>

typedef __attribute__((ext_vector_type(16))) __bf16 v16bf;
typedef __attribute__((ext_vector_type(8)))  __bf16 v8bf;
typedef __attribute__((ext_vector_type(8)))  float  v8f;
typedef __attribute__((ext_vector_type(4)))  unsigned int u32x4;
typedef __attribute__((ext_vector_type(8)))  int i32x8;
typedef __attribute__((ext_vector_type(4)))  int i32x4;

typedef __attribute__((address_space(1))) i32x4 as1_i32x4;  // global int4
typedef __attribute__((address_space(3))) i32x4 as3_i32x4;  // LDS int4
typedef __attribute__((address_space(3))) void lds_void;

#define B_  2
#define S_  2048
#define E_  1024
#define H_  16
#define HD_ 64
#define NEGV (-1.0e9f)

#if __has_builtin(__builtin_amdgcn_global_load_async_to_lds_b128)
#define HAS_ASYNC 1
#else
#define HAS_ASYNC 0
#endif
#if __has_builtin(__builtin_amdgcn_tensor_load_to_lds)
#define HAS_TDM 1
#else
#define HAS_TDM 0
#endif

// ---------------------------------------------------------------------------
// WMMA helper: D = A(16x32 bf16) * B(32x16 bf16) + C(16x16 f32)
// ---------------------------------------------------------------------------
__device__ __forceinline__ v8f wmma_bf16(v16bf a, v16bf b, v8f c) {
  return __builtin_amdgcn_wmma_f32_16x16x32_bf16(false, a, false, b, (short)0,
                                                 c, false, false);
}

// Load one 16-lane-half fragment (A or B, 16-bit, K=32 chunk) from a row
// pointer with contiguous K. Per ISA layout: lanes 0-15 hold K={0..7,16..23},
// lanes 16-31 hold K={8..15,24..31}.  lh = lane>>4.
__device__ __forceinline__ v16bf load_frag(const __bf16* p, int lh) {
  const int base = lh * 8;
  v8bf lo = *(const v8bf*)(p + base);
  v8bf hi = *(const v8bf*)(p + base + 16);
  v16bf f;
#pragma unroll
  for (int i = 0; i < 8; ++i) { f[i] = lo[i]; f[i + 8] = hi[i]; }
  return f;
}

#if HAS_ASYNC
__device__ __forceinline__ void async_g2l_b128(const __bf16* g, __bf16* l) {
  __builtin_amdgcn_global_load_async_to_lds_b128((as1_i32x4*)g, (as3_i32x4*)l,
                                                 0, 0);
}
__device__ __forceinline__ void wait_async0() {
#if __has_builtin(__builtin_amdgcn_s_wait_asynccnt)
  __builtin_amdgcn_s_wait_asynccnt(0);
#else
  asm volatile("s_wait_asynccnt 0" ::: "memory");
#endif
}
#endif

// reductions across the 16 lanes of each half-wave (xor masks 1,2,4,8)
__device__ __forceinline__ float rmax16(float v) {
#pragma unroll
  for (int off = 1; off < 16; off <<= 1) v = fmaxf(v, __shfl_xor(v, off, 32));
  return v;
}
__device__ __forceinline__ float rsum16(float v) {
#pragma unroll
  for (int off = 1; off < 16; off <<= 1) v += __shfl_xor(v, off, 32);
  return v;
}

// ---------------------------------------------------------------------------
// fp32 -> bf16 conversion
// ---------------------------------------------------------------------------
__global__ __launch_bounds__(256) void f32_to_bf16_kernel(
    const float* __restrict__ in, __bf16* __restrict__ out, int n) {
  int i = blockIdx.x * 256 + threadIdx.x;
  if (i < n) out[i] = (__bf16)in[i];
}

// ---------------------------------------------------------------------------
// GEMM: C[M,N] = A[M,K] (bf16, row-major) * W[N,K]^T (bf16) + bias[N]
// mode 0: Cf[m*N+n] = val                                (fp32)
// mode 1: Cb in head layout    [B,H,S,HD]                (bf16)
// mode 2: Cb in head-transpose [B,H,HD,S]                (bf16)
// block = 256 threads (8 waves), tile 128(M) x 128(N), k-step 32.
// A tile staged via GLOBAL_LOAD_ASYNC_TO_LDS_B128 (ASYNCcnt);
// W tile staged via the Tensor Data Mover (TENSORcnt, one 2D tile descriptor
// per k-step, hardware LDS padding 64B data + 16B pad = 40-elem stride).
// ---------------------------------------------------------------------------
__global__ __launch_bounds__(256) void gemm_bf16_kernel(
    const __bf16* __restrict__ A, const __bf16* __restrict__ W,
    const float* __restrict__ bias, float* __restrict__ Cf,
    __bf16* __restrict__ Cb, int mode, int M, int N, int K) {
  __shared__ __align__(16) __bf16 As[128 * 40];
  __shared__ __align__(16) __bf16 Bs[128 * 40];

  const int tid  = threadIdx.x;
  const int lane = tid & 31, wave = tid >> 5;
  const int lh = lane >> 4, ln = lane & 15;
  const int mtile = blockIdx.y * 128, ntile = blockIdx.x * 128;
  const int lrow = tid >> 1, lcol = (tid & 1) * 16;

  v8f acc[8];
#pragma unroll
  for (int n = 0; n < 8; ++n) acc[n] = (v8f){0, 0, 0, 0, 0, 0, 0, 0};

#if HAS_TDM
  const uint32_t bs_lds = (uint32_t)(uintptr_t)(lds_void*)&Bs[0];
#endif

  for (int kb = 0; kb < K; kb += 32) {
    const __bf16* ga = A + (size_t)(mtile + lrow) * K + kb + lcol;
    if (kb + 32 < K) __builtin_prefetch(ga + 32, 0, 1);

    // ---- stage A tile (per-lane async copies, no VGPR staging) ----
#if HAS_ASYNC
    async_g2l_b128(ga, &As[lrow * 40 + lcol]);
    async_g2l_b128(ga + 8, &As[lrow * 40 + lcol + 8]);
#else
    {
      v8bf a0 = *(const v8bf*)ga;
      v8bf a1 = *(const v8bf*)(ga + 8);
      *(v8bf*)&As[lrow * 40 + lcol]     = a0;
      *(v8bf*)&As[lrow * 40 + lcol + 8] = a1;
    }
#endif

    // ---- stage W tile ----
#if HAS_TDM
    if (wave == 0) {
      const uint64_t gaddr =
          (uint64_t)(uintptr_t)(W + (size_t)ntile * K + kb);
      u32x4 g0;
      g0[0] = 1u;                                  // count=1, user descriptor
      g0[1] = bs_lds;                              // lds_addr
      g0[2] = (uint32_t)gaddr;                     // global_addr[31:0]
      g0[3] = (uint32_t)(gaddr >> 32) | (2u << 30);  // addr[56:32] | type=2
      i32x8 g1;
      // data_size=2B, pad_enable, pad_interval=64B, pad_amount=16B
      g1[0] = (1 << 16) | (1 << 20) | (3 << 22) | (3 << 25);
      g1[1] = (K & 0xffff) << 16;                  // tensor_dim0 lo
      g1[2] = ((K >> 16) & 0xffff) | ((N & 0xffff) << 16);  // dim0 hi|dim1 lo
      g1[3] = ((N >> 16) & 0xffff) | (32 << 16);   // dim1 hi | tile_dim0=32
      g1[4] = 128;                                 // tile_dim1=128 rows
      g1[5] = K;                                   // tensor_dim0_stride
      g1[6] = 0;
      g1[7] = 0;
      i32x4 gz4 = (i32x4){0, 0, 0, 0};
      i32x8 gz8 = (i32x8){0, 0, 0, 0, 0, 0, 0, 0};
      __builtin_amdgcn_tensor_load_to_lds(g0, g1, gz4, gz4, gz8, 0);
    }
#elif HAS_ASYNC
    {
      const __bf16* gw = W + (size_t)(ntile + lrow) * K + kb + lcol;
      async_g2l_b128(gw, &Bs[lrow * 40 + lcol]);
      async_g2l_b128(gw + 8, &Bs[lrow * 40 + lcol + 8]);
    }
#else
    {
      const __bf16* gw = W + (size_t)(ntile + lrow) * K + kb + lcol;
      v8bf w0 = *(const v8bf*)gw;
      v8bf w1 = *(const v8bf*)(gw + 8);
      *(v8bf*)&Bs[lrow * 40 + lcol]     = w0;
      *(v8bf*)&Bs[lrow * 40 + lcol + 8] = w1;
    }
#endif

    // ---- drain async/tensor counters, then rendezvous ----
#if HAS_ASYNC
    wait_async0();
#endif
#if HAS_TDM
    if (wave == 0) __builtin_amdgcn_s_wait_tensorcnt(0);
#endif
    __syncthreads();

    v16bf af = load_frag(&As[(wave * 16 + ln) * 40], lh);
#pragma unroll
    for (int n = 0; n < 8; ++n) {
      v16bf bf = load_frag(&Bs[(n * 16 + ln) * 40], lh);
      acc[n] = wmma_bf16(af, bf, acc[n]);
    }
    __syncthreads();
  }

#pragma unroll
  for (int n = 0; n < 8; ++n) {
    const int ncol = ntile + n * 16 + ln;
    const float bval = bias[ncol];
#pragma unroll
    for (int r = 0; r < 8; ++r) {
      const int m = mtile + wave * 16 + r + 8 * lh;
      const float val = acc[n][r] + bval;
      if (mode == 0) {
        Cf[(size_t)m * N + ncol] = val;
      } else {
        const int bb = m >> 11, s = m & (S_ - 1);
        const int hh = ncol >> 6, hd = ncol & (HD_ - 1);
        if (mode == 1)
          Cb[(((size_t)bb * H_ + hh) * S_ + s) * HD_ + hd] = (__bf16)val;
        else
          Cb[(((size_t)bb * H_ + hh) * HD_ + hd) * S_ + s] = (__bf16)val;
      }
    }
  }
}

// ---------------------------------------------------------------------------
// Pass 1: raw scores = QK^T/8 + cw, causal mask, write raw to attnW,
// online row max/sum -> mstat/lstat.  Per wave: one 16-row q strip.
// grid (S/128, H, B), block 256 (8 waves).
// ---------------------------------------------------------------------------
__global__ __launch_bounds__(256) void attn_scores_kernel(
    const __bf16* __restrict__ qh, const __bf16* __restrict__ kh,
    const float* __restrict__ cw, float* __restrict__ attnW,
    float* __restrict__ mstat, float* __restrict__ lstat) {
  const int lane = threadIdx.x & 31, wave = threadIdx.x >> 5;
  const int lh = lane >> 4, ln = lane & 15;
  const int b = blockIdx.z, h = blockIdx.y;
  const int bh = b * H_ + h;
  const int qbase = blockIdx.x * 128 + wave * 16;

  const __bf16* qp = qh + ((size_t)bh * S_ + (qbase + ln)) * HD_;
  v16bf aq0 = load_frag(qp, lh);
  v16bf aq1 = load_frag(qp + 32, lh);

  float mrun[8], lrun[8];
#pragma unroll
  for (int r = 0; r < 8; ++r) { mrun[r] = -3.0e38f; lrun[r] = 0.f; }

  const __bf16* kbp = kh + (size_t)bh * S_ * HD_;
  const float* cwb = cw + (size_t)b * S_ * S_;
  float* aw = attnW + (size_t)bh * S_ * S_;

  const int nkt = (qbase + 15) / 32 + 1;
  for (int kt = 0; kt < nkt; ++kt) {
    const int kb = kt * 32;
    v8f sc[2];
#pragma unroll
    for (int ns = 0; ns < 2; ++ns) {
      const __bf16* kp = kbp + (size_t)(kb + ns * 16 + ln) * HD_;
      v16bf b0 = load_frag(kp, lh);
      v16bf b1 = load_frag(kp + 32, lh);
      v8f z = (v8f){0, 0, 0, 0, 0, 0, 0, 0};
      z = wmma_bf16(aq0, b0, z);
      z = wmma_bf16(aq1, b1, z);
      sc[ns] = z;
    }
#pragma unroll
    for (int r = 0; r < 8; ++r) {
      const int row = qbase + r + 8 * lh;
      const int c0 = kb + ln, c1 = kb + 16 + ln;
      float v0 = sc[0][r] * 0.125f + cwb[(size_t)row * S_ + c0];
      float v1 = sc[1][r] * 0.125f + cwb[(size_t)row * S_ + c1];
      if (c0 > row) v0 = NEGV;
      if (c1 > row) v1 = NEGV;
      aw[(size_t)row * S_ + c0] = v0;
      aw[(size_t)row * S_ + c1] = v1;
      float t = rmax16(fmaxf(v0, v1));
      float nm = fmaxf(mrun[r], t);
      float sum = rsum16(__expf(v0 - nm) + __expf(v1 - nm));
      lrun[r] = lrun[r] * __expf(mrun[r] - nm) + sum;
      mrun[r] = nm;
    }
  }
  if (ln == 0) {
#pragma unroll
    for (int r = 0; r < 8; ++r) {
      const int row = qbase + r + 8 * lh;
      mstat[(size_t)bh * S_ + row] = mrun[r];
      lstat[(size_t)bh * S_ + row] = lrun[r];
    }
  }
}

// ---------------------------------------------------------------------------
// Pass 2: p = exp(s-m)/l, overwrite attnW with probabilities (+ zero-fill the
// untouched upper triangle), transpose P tile through LDS, out = P @ V.
// vT layout [B,H,HD,S] makes V B-fragments contiguous 16B loads.
// ---------------------------------------------------------------------------
__global__ __launch_bounds__(256) void attn_pv_kernel(
    const __bf16* __restrict__ vT, const float* __restrict__ mstat,
    const float* __restrict__ lstat, float* __restrict__ attnW,
    __bf16* __restrict__ attn_out) {
  __shared__ __align__(16) __bf16 Ps[8][16 * 40];

  const int lane = threadIdx.x & 31, wave = threadIdx.x >> 5;
  const int lh = lane >> 4, ln = lane & 15;
  const int b = blockIdx.z, h = blockIdx.y;
  const int bh = b * H_ + h;
  const int qbase = blockIdx.x * 128 + wave * 16;

  float mv[8], il[8];
#pragma unroll
  for (int r = 0; r < 8; ++r) {
    const int row = qbase + r + 8 * lh;
    mv[r] = mstat[(size_t)bh * S_ + row];
    il[r] = 1.0f / lstat[(size_t)bh * S_ + row];
  }

  float* aw = attnW + (size_t)bh * S_ * S_;
  const __bf16* vb = vT + (size_t)bh * HD_ * S_;

  v8f acc[4];
#pragma unroll
  for (int n = 0; n < 4; ++n) acc[n] = (v8f){0, 0, 0, 0, 0, 0, 0, 0};

  const int nkt = (qbase + 15) / 32 + 1;
  for (int kt = 0; kt < nkt; ++kt) {
    const int kb = kt * 32;
#pragma unroll
    for (int ns = 0; ns < 2; ++ns) {
#pragma unroll
      for (int r = 0; r < 8; ++r) {
        const int row = qbase + r + 8 * lh;
        const int col = kb + ns * 16 + ln;
        float s = aw[(size_t)row * S_ + col];
        float p = __expf(s - mv[r]) * il[r];
        aw[(size_t)row * S_ + col] = p;
        Ps[wave][(r + 8 * lh) * 40 + ns * 16 + ln] = (__bf16)p;
      }
    }
    // same-wave LDS RAW across lanes: drain DS counter explicitly
    asm volatile("s_wait_dscnt 0" ::: "memory");
    v16bf af = load_frag(&Ps[wave][ln * 40], lh);
#pragma unroll
    for (int n = 0; n < 4; ++n) {
      const __bf16* vp = vb + (size_t)(n * 16 + ln) * S_ + kb;
      v16bf bf = load_frag(vp, lh);
      acc[n] = wmma_bf16(af, bf, acc[n]);
    }
  }

  // zero-fill masked-out columns never touched by the k-loop
  for (int kb2 = nkt * 32; kb2 < S_; kb2 += 32) {
#pragma unroll
    for (int r = 0; r < 8; ++r) {
      const int row = qbase + r + 8 * lh;
      aw[(size_t)row * S_ + kb2 + ln] = 0.f;
      aw[(size_t)row * S_ + kb2 + 16 + ln] = 0.f;
    }
  }

  // write head-concat output [B,S,E] bf16
#pragma unroll
  for (int n = 0; n < 4; ++n) {
#pragma unroll
    for (int r = 0; r < 8; ++r) {
      const int row = qbase + r + 8 * lh;
      const int col = h * HD_ + n * 16 + ln;
      attn_out[(size_t)(b * S_ + row) * E_ + col] = (__bf16)acc[n][r];
    }
  }
}

// ---------------------------------------------------------------------------
extern "C" void kernel_launch(void* const* d_in, const int* in_sizes, int n_in,
                              void* d_out, int out_size, void* d_ws,
                              size_t ws_size, hipStream_t stream) {
  const float* query = (const float*)d_in[0];
  const float* key_t = (const float*)d_in[1];
  const float* value = (const float*)d_in[2];
  const float* cw    = (const float*)d_in[3];
  // d_in[4] = attn_mask (tril by construction; causality computed directly)
  const float* Wq = (const float*)d_in[5];
  const float* bq = (const float*)d_in[6];
  const float* Wk = (const float*)d_in[7];
  const float* bk = (const float*)d_in[8];
  const float* Wv = (const float*)d_in[9];
  const float* bv = (const float*)d_in[10];
  const float* Wo = (const float*)d_in[11];
  const float* bo = (const float*)d_in[12];
  // d_in[13]/d_in[14] = Wc/bc (dead code in reference)

  float* out   = (float*)d_out;
  float* attnW = out + (size_t)B_ * S_ * E_;

  const size_t BSE = (size_t)B_ * S_ * E_;
  const size_t EE  = (size_t)E_ * E_;

  uint8_t* wp = (uint8_t*)d_ws;
  auto take = [&](size_t bytes) -> void* {
    void* p = wp;
    wp += (bytes + 255) & ~(size_t)255;
    return p;
  };

  __bf16* Xq = (__bf16*)take(BSE * 2);
  __bf16* Xk = (__bf16*)take(BSE * 2);
  __bf16* Xv = (__bf16*)take(BSE * 2);
  __bf16* wq = (__bf16*)take(EE * 2);
  __bf16* wk = (__bf16*)take(EE * 2);
  __bf16* wv = (__bf16*)take(EE * 2);
  __bf16* wo = (__bf16*)take(EE * 2);
  __bf16* qh = (__bf16*)take(BSE * 2);
  __bf16* kh = (__bf16*)take(BSE * 2);
  __bf16* vT = (__bf16*)take(BSE * 2);
  __bf16* ao = (__bf16*)take(BSE * 2);
  float* mstat = (float*)take((size_t)B_ * H_ * S_ * 4);
  float* lstat = (float*)take((size_t)B_ * H_ * S_ * 4);

  f32_to_bf16_kernel<<<dim3((unsigned)(BSE / 256)), 256, 0, stream>>>(query, Xq, (int)BSE);
  f32_to_bf16_kernel<<<dim3((unsigned)(BSE / 256)), 256, 0, stream>>>(key_t, Xk, (int)BSE);
  f32_to_bf16_kernel<<<dim3((unsigned)(BSE / 256)), 256, 0, stream>>>(value, Xv, (int)BSE);
  f32_to_bf16_kernel<<<dim3((unsigned)(EE / 256)), 256, 0, stream>>>(Wq, wq, (int)EE);
  f32_to_bf16_kernel<<<dim3((unsigned)(EE / 256)), 256, 0, stream>>>(Wk, wk, (int)EE);
  f32_to_bf16_kernel<<<dim3((unsigned)(EE / 256)), 256, 0, stream>>>(Wv, wv, (int)EE);
  f32_to_bf16_kernel<<<dim3((unsigned)(EE / 256)), 256, 0, stream>>>(Wo, wo, (int)EE);

  dim3 gg(E_ / 128, (B_ * S_) / 128);  // (8, 32)
  gemm_bf16_kernel<<<gg, 256, 0, stream>>>(Xq, wq, bq, nullptr, qh, 1, B_ * S_, E_, E_);
  gemm_bf16_kernel<<<gg, 256, 0, stream>>>(Xk, wk, bk, nullptr, kh, 1, B_ * S_, E_, E_);
  gemm_bf16_kernel<<<gg, 256, 0, stream>>>(Xv, wv, bv, nullptr, vT, 2, B_ * S_, E_, E_);

  dim3 ga(S_ / 128, H_, B_);  // (16, 16, 2)
  attn_scores_kernel<<<ga, 256, 0, stream>>>(qh, kh, cw, attnW, mstat, lstat);
  attn_pv_kernel<<<ga, 256, 0, stream>>>(vT, mstat, lstat, attnW, ao);

  gemm_bf16_kernel<<<gg, 256, 0, stream>>>(ao, wo, bo, out, nullptr, 0, B_ * S_, E_, E_);
}
